// TransformerBlock_1082331759111
// MI455X (gfx1250) — matrix operand order repaired and checked by reading, not compiled, not run on hardware
//
#include <hip/hip_runtime.h>

#define NTOK   2048
#define DMODEL 1024
#define NHEAD  16
#define DHEAD  64
#define FFDIM  4096
#define LNEPS  1e-5f

typedef __attribute__((ext_vector_type(16))) __bf16 v16bf;
typedef __attribute__((ext_vector_type(8)))  float  v8f;
typedef unsigned short u16b;

union Frag32B { uint4 u[2]; v16bf v; };

__device__ __forceinline__ u16b f2bf(float x) {
  union { float f; unsigned u; } t; t.f = x;
  unsigned r = t.u + 0x7fffu + ((t.u >> 16) & 1u);   // round-to-nearest-even
  return (u16b)(r >> 16);
}

// Load a 16-element bf16 fragment as two contiguous 16B chunks from one row.
__device__ __forceinline__ v16bf load_frag(const u16b* rowbase, int off0, int off1) {
  Frag32B f;
  f.u[0] = *(const uint4*)(rowbase + off0);
  f.u[1] = *(const uint4*)(rowbase + off1);
  return f.v;
}

__device__ __forceinline__ v8f wmma_bf(v16bf a, v16bf b, v8f c) {
  return __builtin_amdgcn_wmma_f32_16x16x32_bf16(false, a, false, b, (short)0, c,
                                                 false, false);
}

// --- CDNA5 async global->LDS copy (ASYNCcnt-tracked) ----------------------
__device__ __forceinline__ void async_b128(unsigned lds_off, const u16b* gaddr) {
  asm volatile("global_load_async_to_lds_b128 %0, %1, off"
               :: "v"(lds_off), "v"((unsigned long long)(uintptr_t)gaddr)
               : "memory");
}

template <int N>
__device__ __forceinline__ void wait_asynccnt() {
#if __has_builtin(__builtin_amdgcn_s_wait_asynccnt)
  __builtin_amdgcn_s_wait_asynccnt(N);
#else
  asm volatile("s_wait_asynccnt %0" :: "n"(N) : "memory");
#endif
}

// ---------------------------------------------------------------------------
// Elementwise fp32 -> bf16 convert
// ---------------------------------------------------------------------------
__global__ void cvt_bf16_kernel(const float* __restrict__ src, u16b* __restrict__ dst, long n) {
  long i = (long)blockIdx.x * blockDim.x + threadIdx.x;
  if (i < n) dst[i] = f2bf(src[i]);
}

// ---------------------------------------------------------------------------
// Generic transpose+convert: dst[j*I + i] = bf16( src[(j/sub)*s_hi + i*s_in + (j%sub)*s_lo] )
// Produces Bt layout [J(out), I(K)] bf16 row-major.
// ---------------------------------------------------------------------------
__global__ void transpose_cvt_kernel(const float* __restrict__ src, u16b* __restrict__ dst,
                                     int J, int I, int sub, long s_hi, long s_in, long s_lo) {
  long idx = (long)blockIdx.x * blockDim.x + threadIdx.x;
  long total = (long)J * I;
  if (idx >= total) return;
  int j = (int)(idx / I);
  int i = (int)(idx % I);
  float v = src[(long)(j / sub) * s_hi + (long)i * s_in + (long)(j % sub) * s_lo];
  dst[idx] = f2bf(v);
}

// ---------------------------------------------------------------------------
// Tiled WMMA GEMM: C[M,Nout] = A[M,K](bf16) * Bt[Nout,K](bf16)^T, fp32 accum.
// Block = 128 threads = 4 waves. Block tile 128x64; each wave owns 32x64
// (8 accumulators). B tile (64 cols x 32 K) is staged into LDS once per
// block per K-step via global_load_async_to_lds_b128, double-buffered.
// ---------------------------------------------------------------------------
enum { MODE_QKV = 0, MODE_WO = 1, MODE_FF1 = 2, MODE_FF2 = 3 };

__device__ __forceinline__ void stage_B_async(const u16b* __restrict__ Bt, int K,
                                              int cbase, int kk, u16b* buf) {
  // 64 rows x 32 K-elems = 256 chunks of 16B; 128 threads -> 2 chunks each.
  const int c0 = threadIdx.x;
#pragma unroll
  for (int j = 0; j < 2; ++j) {
    int c   = c0 + j * 128;
    int row = c >> 2;
    int kc  = (c & 3) * 8;
    const u16b* g = Bt + (size_t)(cbase + row) * K + kk + kc;
    unsigned lds_off = (unsigned)(unsigned long long)(uintptr_t)(buf + (size_t)c * 8);
    async_b128(lds_off, g);
  }
}

template <int MODE>
__global__ __launch_bounds__(128) void gemm_kernel(
    const u16b* __restrict__ A, const u16b* __restrict__ Bt,
    int M, int Nout, int K,
    const float* __restrict__ bias,    // [Nout] or null
    const float* __restrict__ resid,   // fp32 [M,Nout] or null
    float* __restrict__ outf,          // fp32 [M,Nout] or null
    u16b* __restrict__ outb,           // bf16 [M,Nout] or null
    u16b* __restrict__ qb, u16b* __restrict__ kb_o, u16b* __restrict__ vtb) {
  __shared__ __align__(16) u16b Bs[2][64][32];   // 8 KB double buffer

  const int lane = threadIdx.x & 31;
  const int wave = threadIdx.x >> 5;
  const int half = lane >> 4;
  const int l15  = lane & 15;
  const int r0    = blockIdx.x * 128 + wave * 32;
  const int cbase = blockIdx.y * 64;

  const v8f zero = {0.f, 0.f, 0.f, 0.f, 0.f, 0.f, 0.f, 0.f};
  v8f acc[8];
#pragma unroll
  for (int t = 0; t < 8; ++t) acc[t] = zero;

  const u16b* arow0 = A + (size_t)(r0 + l15) * K;
  const u16b* arow1 = A + (size_t)(r0 + 16 + l15) * K;

  stage_B_async(Bt, K, cbase, 0, &Bs[0][0][0]);
  int cur = 0;
  for (int kk = 0; kk < K; kk += 32) {
    if (kk + 32 < K) {
      stage_B_async(Bt, K, cbase, kk + 32, &Bs[cur ^ 1][0][0]);
      wait_asynccnt<2>();   // the 2 just-issued may remain outstanding
    } else {
      wait_asynccnt<0>();
    }
    __syncthreads();        // current buffer visible to all waves

    v16bf a0 = load_frag(arow0, kk + half * 8, kk + 16 + half * 8);
    v16bf a1 = load_frag(arow1, kk + half * 8, kk + 16 + half * 8);
#pragma unroll
    for (int t = 0; t < 4; ++t) {
      Frag32B bf;
      bf.u[0] = *(const uint4*)&Bs[cur][t * 16 + l15][half * 8];
      bf.u[1] = *(const uint4*)&Bs[cur][t * 16 + l15][16 + half * 8];
      acc[t]     = wmma_bf(a0, bf.v, acc[t]);
      acc[4 + t] = wmma_bf(a1, bf.v, acc[4 + t]);
    }
    __syncthreads();        // done reading before this buffer is re-staged
    cur ^= 1;
  }

#pragma unroll
  for (int rg = 0; rg < 2; ++rg) {
#pragma unroll
    for (int t = 0; t < 4; ++t) {
#pragma unroll
      for (int i = 0; i < 8; ++i) {
        int row = r0 + rg * 16 + i + 8 * half;
        int col = cbase + t * 16 + l15;
        float v = acc[rg * 4 + t][i];
        if (MODE == MODE_QKV) {
          int sel = col >> 10;          // 0:q 1:k 2:v
          int c2  = col & 1023;
          int h   = c2 >> 6;
          int dk  = c2 & 63;
          if (sel == 0)      qb[((size_t)h * NTOK + row) * DHEAD + dk] = f2bf(v);
          else if (sel == 1) kb_o[((size_t)h * NTOK + row) * DHEAD + dk] = f2bf(v);
          else               vtb[((size_t)h * DHEAD + dk) * NTOK + row] = f2bf(v);
        } else if (MODE == MODE_WO) {
          outf[(size_t)row * Nout + col] = v + resid[(size_t)row * Nout + col];
        } else if (MODE == MODE_FF1) {
          v += bias[col];
          v = v > 0.f ? v : 0.f;
          outb[(size_t)row * Nout + col] = f2bf(v);
        } else {  // MODE_FF2: relu(acc+b2) + h
          v += bias[col];
          v = v > 0.f ? v : 0.f;
          outf[(size_t)row * Nout + col] = v + resid[(size_t)row * Nout + col];
        }
      }
    }
  }
}

// ---------------------------------------------------------------------------
// Flash attention: one wave per (16-query block, head). Online softmax over
// 32-key blocks. q/k bf16 [H,N,64]; v transposed bf16 [H,64,N].
// Output bf16 concat [N, H*64].
// ---------------------------------------------------------------------------
__global__ __launch_bounds__(32) void attention_kernel(
    const u16b* __restrict__ qb, const u16b* __restrict__ kb,
    const u16b* __restrict__ vtb, u16b* __restrict__ concat) {
  __shared__ __align__(16) u16b Pl[16][32];

  const int lane = threadIdx.x & 31;
  const int half = lane >> 4;
  const int l15  = lane & 15;
  const int q0   = blockIdx.x * 16;
  const int h    = blockIdx.y;

  const u16b* qh = qb + (size_t)h * NTOK * DHEAD;
  const u16b* kh = kb + (size_t)h * NTOK * DHEAD;
  const u16b* vh = vtb + (size_t)h * DHEAD * NTOK;

  // Persistent Q fragments: rows q0..q0+15, DK 0..31 and 32..63.
  const u16b* qrow = qh + (size_t)(q0 + l15) * DHEAD;
  v16bf qf0 = load_frag(qrow, half * 16, half * 16 + 8);
  v16bf qf1 = load_frag(qrow, 32 + half * 16, 32 + half * 16 + 8);

  const v8f zero = {0.f, 0.f, 0.f, 0.f, 0.f, 0.f, 0.f, 0.f};
  v8f acc[4] = {zero, zero, zero, zero};
  float Mrow[8], Lrow[8];
#pragma unroll
  for (int i = 0; i < 8; ++i) { Mrow[i] = -1e30f; Lrow[i] = 0.f; }

  for (int kb0 = 0; kb0 <= q0; kb0 += 32) {
    // Two 16x16 score tiles covering keys [kb0, kb0+32)
    v8f s01[2];
#pragma unroll
    for (int tt = 0; tt < 2; ++tt) {
      const u16b* krow = kh + (size_t)(kb0 + tt * 16 + l15) * DHEAD;
      v16bf kf0 = load_frag(krow, half * 16, half * 16 + 8);
      v16bf kf1 = load_frag(krow, 32 + half * 16, 32 + half * 16 + 8);
      v8f s = zero;
      s = wmma_bf(qf0, kf0, s);
      s = wmma_bf(qf1, kf1, s);
#pragma unroll
      for (int i = 0; i < 8; ++i) {
        int qg = q0 + i + 8 * half;
        int kg = kb0 + tt * 16 + l15;
        float sv = s[i] * 0.125f;               // 1/sqrt(64)
        s[i] = (kg <= qg) ? sv : -1e30f;        // causal mask
      }
      s01[tt] = s;
    }

    // Online softmax update (row m = i + 8*half lives in lanes of one half)
#pragma unroll
    for (int i = 0; i < 8; ++i) {
      float mx = fmaxf(s01[0][i], s01[1][i]);
      for (int off = 1; off < 16; off <<= 1) mx = fmaxf(mx, __shfl_xor(mx, off, 32));
      float newM = fmaxf(Mrow[i], mx);
      float sc = __expf(Mrow[i] - newM);
      Mrow[i] = newM;
      Lrow[i] *= sc;
#pragma unroll
      for (int t = 0; t < 4; ++t) acc[t][i] *= sc;
      float p0 = __expf(s01[0][i] - newM);
      float p1 = __expf(s01[1][i] - newM);
      float rs = p0 + p1;
      for (int off = 1; off < 16; off <<= 1) rs += __shfl_xor(rs, off, 32);
      Lrow[i] += rs;
      Pl[i + 8 * half][l15]      = f2bf(p0);
      Pl[i + 8 * half][16 + l15] = f2bf(p1);
    }
    __syncthreads();
    // Re-gather P as a 16x32 A-fragment
    Frag32B pf;
    pf.u[0] = *(const uint4*)&Pl[l15][half * 16];
    pf.u[1] = *(const uint4*)&Pl[l15][half * 16 + 8];
    __syncthreads();

    // acc += P @ V   (V from transposed vt: contiguous keys per (dv) row)
#pragma unroll
    for (int t = 0; t < 4; ++t) {
      const u16b* vrow = vh + (size_t)(t * 16 + l15) * NTOK;
      v16bf vf = load_frag(vrow, kb0 + half * 16, kb0 + half * 16 + 8);
      acc[t] = wmma_bf(pf.v, vf, acc[t]);
    }
  }

  // Normalize and store into concat [N, H*64] (head-order concat)
#pragma unroll
  for (int t = 0; t < 4; ++t) {
#pragma unroll
    for (int i = 0; i < 8; ++i) {
      int row = q0 + i + 8 * half;
      int col = h * DHEAD + t * 16 + l15;
      concat[(size_t)row * (NHEAD * DHEAD) + col] = f2bf(acc[t][i] / Lrow[i]);
    }
  }
}

// ---------------------------------------------------------------------------
// Row LayerNorm over D=1024. Optionally also writes bf16 copy.
// ---------------------------------------------------------------------------
template <bool WRITE_BF>
__global__ __launch_bounds__(256) void layernorm_kernel(
    const float* __restrict__ y, const float* __restrict__ g, const float* __restrict__ b,
    float* __restrict__ outf, u16b* __restrict__ outb) {
  __shared__ float red[256];
  const int row = blockIdx.x;
  const float* yr = y + (size_t)row * DMODEL;

  float s = 0.f;
  for (int i = threadIdx.x; i < DMODEL; i += 256) s += yr[i];
  red[threadIdx.x] = s;
  __syncthreads();
  for (int off = 128; off > 0; off >>= 1) {
    if ((int)threadIdx.x < off) red[threadIdx.x] += red[threadIdx.x + off];
    __syncthreads();
  }
  float mu = red[0] * (1.0f / DMODEL);
  __syncthreads();

  float s2 = 0.f;
  for (int i = threadIdx.x; i < DMODEL; i += 256) {
    float d = yr[i] - mu;
    s2 += d * d;
  }
  red[threadIdx.x] = s2;
  __syncthreads();
  for (int off = 128; off > 0; off >>= 1) {
    if ((int)threadIdx.x < off) red[threadIdx.x] += red[threadIdx.x + off];
    __syncthreads();
  }
  float rstd = rsqrtf(red[0] * (1.0f / DMODEL) + LNEPS);

  for (int i = threadIdx.x; i < DMODEL; i += 256) {
    float hv = (yr[i] - mu) * rstd * g[i] + b[i];
    outf[(size_t)row * DMODEL + i] = hv;
    if (WRITE_BF) outb[(size_t)row * DMODEL + i] = f2bf(hv);
  }
}

// ---------------------------------------------------------------------------
extern "C" void kernel_launch(void* const* d_in, const int* in_sizes, int n_in,
                              void* d_out, int out_size, void* d_ws, size_t ws_size,
                              hipStream_t stream) {
  (void)in_sizes; (void)n_in; (void)out_size; (void)ws_size;
  const float* x    = (const float*)d_in[0];
  const float* w_q  = (const float*)d_in[1];
  const float* w_k  = (const float*)d_in[2];
  const float* w_v  = (const float*)d_in[3];
  const float* w_o  = (const float*)d_in[4];
  const float* ln1g = (const float*)d_in[5];
  const float* ln1b = (const float*)d_in[6];
  const float* w1   = (const float*)d_in[7];
  const float* b1   = (const float*)d_in[8];
  const float* w2   = (const float*)d_in[9];
  const float* b2   = (const float*)d_in[10];
  const float* ln2g = (const float*)d_in[11];
  const float* ln2b = (const float*)d_in[12];

  char* ws = (char*)d_ws;
  size_t off = 0;
  auto carve = [&](size_t bytes) -> char* {
    char* p = ws + off;
    off += (bytes + 255) & ~(size_t)255;
    return p;
  };

  u16b* x_bf    = (u16b*)carve((size_t)NTOK * DMODEL * 2);
  u16b* wqkv_t  = (u16b*)carve((size_t)3 * DMODEL * DMODEL * 2);  // [3072,1024]
  u16b* wo_t    = (u16b*)carve((size_t)DMODEL * DMODEL * 2);
  u16b* w1_t    = (u16b*)carve((size_t)FFDIM * DMODEL * 2);
  u16b* w2_t    = (u16b*)carve((size_t)DMODEL * FFDIM * 2);
  u16b* q_bf    = (u16b*)carve((size_t)NHEAD * NTOK * DHEAD * 2);
  u16b* k_bf    = (u16b*)carve((size_t)NHEAD * NTOK * DHEAD * 2);
  u16b* vt_bf   = (u16b*)carve((size_t)NHEAD * DHEAD * NTOK * 2);
  u16b* cc_bf   = (u16b*)carve((size_t)NTOK * DMODEL * 2);
  float* y1     = (float*)carve((size_t)NTOK * DMODEL * 4);
  float* h_f    = (float*)carve((size_t)NTOK * DMODEL * 4);
  u16b* h_bf    = (u16b*)carve((size_t)NTOK * DMODEL * 2);
  u16b* ff1_bf  = (u16b*)carve((size_t)NTOK * FFDIM * 2);
  float* y2     = (float*)carve((size_t)NTOK * DMODEL * 4);

  // --- operand conversion / weight transposition (bf16, [out,K] layout) ---
  {
    long n = (long)NTOK * DMODEL;
    cvt_bf16_kernel<<<(int)((n + 255) / 256), 256, 0, stream>>>(x, x_bf, n);
  }
  int tc1 = (DMODEL * DMODEL + 255) / 256;
  // w_q/w_k/w_v: [H,D,64] -> Bt[h*64+k][d]
  transpose_cvt_kernel<<<tc1, 256, 0, stream>>>(w_q, wqkv_t, 1024, 1024, 64,
                                                (long)DMODEL * 64, 64, 1);
  transpose_cvt_kernel<<<tc1, 256, 0, stream>>>(w_k, wqkv_t + (size_t)DMODEL * DMODEL,
                                                1024, 1024, 64, (long)DMODEL * 64, 64, 1);
  transpose_cvt_kernel<<<tc1, 256, 0, stream>>>(w_v, wqkv_t + (size_t)2 * DMODEL * DMODEL,
                                                1024, 1024, 64, (long)DMODEL * 64, 64, 1);
  // w_o [1024,1024] (in,out) -> Bt[out][in]
  transpose_cvt_kernel<<<tc1, 256, 0, stream>>>(w_o, wo_t, 1024, 1024, 1024, 0, 1024, 1);
  // w1 [1024,4096] -> Bt[4096][1024]
  transpose_cvt_kernel<<<(FFDIM * DMODEL + 255) / 256, 256, 0, stream>>>(
      w1, w1_t, FFDIM, DMODEL, FFDIM, 0, FFDIM, 1);
  // w2 [4096,1024] -> Bt[1024][4096]
  transpose_cvt_kernel<<<(FFDIM * DMODEL + 255) / 256, 256, 0, stream>>>(
      w2, w2_t, DMODEL, FFDIM, DMODEL, 0, DMODEL, 1);

  // --- fused QKV projection: [2048,3072] = x_bf @ wqkv_t^T ---
  {
    dim3 g(NTOK / 128, (3 * DMODEL) / 64);
    gemm_kernel<MODE_QKV><<<g, 128, 0, stream>>>(x_bf, wqkv_t, NTOK, 3 * DMODEL, DMODEL,
                                                 nullptr, nullptr, nullptr, nullptr,
                                                 q_bf, k_bf, vt_bf);
  }

  // --- causal flash attention ---
  {
    dim3 g(NTOK / 16, NHEAD);
    attention_kernel<<<g, 32, 0, stream>>>(q_bf, k_bf, vt_bf, cc_bf);
  }

  // --- output projection + residual: y1 = x + concat @ w_o ---
  {
    dim3 g(NTOK / 128, DMODEL / 64);
    gemm_kernel<MODE_WO><<<g, 128, 0, stream>>>(cc_bf, wo_t, NTOK, DMODEL, DMODEL,
                                                nullptr, x, y1, nullptr,
                                                nullptr, nullptr, nullptr);
  }
  layernorm_kernel<true><<<NTOK, 256, 0, stream>>>(y1, ln1g, ln1b, h_f, h_bf);

  // --- FFN1: ff1 = relu(h @ w1 + b1) (bf16) ---
  {
    dim3 g(NTOK / 128, FFDIM / 64);
    gemm_kernel<MODE_FF1><<<g, 128, 0, stream>>>(h_bf, w1_t, NTOK, FFDIM, DMODEL,
                                                 b1, nullptr, nullptr, ff1_bf,
                                                 nullptr, nullptr, nullptr);
  }
  // --- FFN2: y2 = h + relu(ff1 @ w2 + b2) ---
  {
    dim3 g(NTOK / 128, DMODEL / 64);
    gemm_kernel<MODE_FF2><<<g, 128, 0, stream>>>(ff1_bf, w2_t, NTOK, DMODEL, FFDIM,
                                                 b2, h_f, y2, nullptr,
                                                 nullptr, nullptr, nullptr);
  }
  layernorm_kernel<false><<<NTOK, 256, 0, stream>>>(y2, ln2g, ln2b, (float*)d_out, nullptr);
}
